// GIN_83872121356545
// MI455X (gfx1250) — compile-verified, matrix operand order checked by reading, not run on hardware
//
#include <hip/hip_runtime.h>
#include <hip/hip_bf16.h>

// ---------------------------------------------------------------------------
// GIN 4-layer forward for MI455X (gfx1250, wave32, WMMA bf16 16x16x32).
//   zin = h + scatter_add(h[src] -> dst)             (f32 atomics)
//   z1  = zin @ W1^T    (WMMA bf16, BN col-stats fused in epilogue)
//   BN finalize -> scale/shift
//   z2  = SELU(BN(z1)) @ W2^T + b2 (WMMA bf16; BN+SELU fused into LDS staging;
//                                   per-graph sums fused in epilogue)
//   GraphNorm (two passes w/ atomics) + SELU -> h, pooled accumulation
// Final: pooled /= cnt.
// GEMM blocking: 32 rows/block (2 M-tiles per wave) to halve L2 weight
// traffic and issue two WMMAs per B fragment.
// ---------------------------------------------------------------------------

typedef __attribute__((ext_vector_type(16))) __bf16 v16bf;
typedef __attribute__((ext_vector_type(8)))  __bf16 v8bf;
typedef __attribute__((ext_vector_type(8)))  float  v8f;

#define Dk   128   // node feature dim
#define FH   512   // hidden (4*H)
#define NG   256   // graphs
#define BN_EPS 1e-5f
#define GN_EPS 1e-5f

__device__ __forceinline__ float selu_f(float x) {
    const float lam = 1.0507009873554805f;
    const float alp = 1.6732632423543772f;
    return x > 0.0f ? lam * x : lam * alp * (__expf(x) - 1.0f);
}

// Build a 16-element bf16 WMMA fragment from two 8-element chunks at p and p+16.
// Per CDNA5 ISA 16-bit A 16x32 layout: lane's vector elems 0..7 = K base..base+7,
// elems 8..15 = K base+16..base+23 (base = (lane>=16) ? 8 : 0, plus k-step).
__device__ __forceinline__ v16bf frag16(const __bf16* p) {
    v8bf c0 = *(const v8bf*)p;
    v8bf c1 = *(const v8bf*)(p + 16);
    return __builtin_shufflevector(c0, c1, 0,1,2,3,4,5,6,7,8,9,10,11,12,13,14,15);
}

// ------------------------------ utility kernels -----------------------------

__global__ void k_zero(float* __restrict__ p, int n) {
    int i = blockIdx.x * blockDim.x + threadIdx.x;
    if (i < n) p[i] = 0.0f;
}

__global__ void k_cvt_bf16(const float* __restrict__ s, __bf16* __restrict__ d, int n) {
    int i = blockIdx.x * blockDim.x + threadIdx.x;
    if (i < n) d[i] = (__bf16)s[i];
}

__global__ void k_copy(const float* __restrict__ s, float* __restrict__ d, int n) {
    int i = blockIdx.x * blockDim.x + threadIdx.x;
    if (i < n) d[i] = s[i];
}

__global__ void k_count(const int* __restrict__ batch, float* __restrict__ cnt, int n) {
    int i = blockIdx.x * blockDim.x + threadIdx.x;
    if (i < n) atomicAdd(&cnt[batch[i]], 1.0f);
}

// ----------------------------- edge scatter-add -----------------------------
__global__ void k_scatter(const float* __restrict__ h, const int* __restrict__ ei,
                          float* __restrict__ zin, int E) {
    int idx = blockIdx.x * blockDim.x + threadIdx.x;
    if (idx >= E * 32) return;
    int e = idx >> 5, q = (idx & 31) << 2;
    int s = ei[e];
    int t = ei[E + e];
    float4 v = *(const float4*)(h + (size_t)s * Dk + q);
    float* o = zin + (size_t)t * Dk + q;
    atomicAdd(o + 0, v.x);
    atomicAdd(o + 1, v.y);
    atomicAdd(o + 2, v.z);
    atomicAdd(o + 3, v.w);
}

// ------------------------- GEMM1: [N,128] x [128,512] -----------------------
// 256 threads = 8 waves; block owns 32 rows (2 M-tiles); wave w covers cols
// w*64..w*64+63 (4 N-tiles); K loop 4 x 32. Each B fragment feeds 2 WMMAs.
// BN column sums/sumsq fused in epilogue.
__global__ void __launch_bounds__(256)
k_gemm1(const float* __restrict__ zin, const __bf16* __restrict__ Wb /*[512][128]*/,
        __bf16* __restrict__ z1, float* __restrict__ colsum, float* __restrict__ colsq,
        int Nn) {
    __shared__ __bf16 At[32 * Dk];   // 8 KB, row-major [32][128]
    const int tid  = threadIdx.x;
    const int row0 = blockIdx.x * 32;

    // Stage A tile (f32 -> bf16). 1024 float4 total, 4 per thread.
    for (int i = tid; i < 1024; i += 256) {
        int r = i >> 5;                    // 32 float4 per row
        int b = i << 2;                    // flat bf16 index = r*128 + (i&31)*4
        if (row0 + r < Nn) {
            float4 v = *(const float4*)(zin + (size_t)(row0 + r) * Dk + ((i & 31) << 2));
            At[b] = (__bf16)v.x; At[b+1] = (__bf16)v.y;
            At[b+2] = (__bf16)v.z; At[b+3] = (__bf16)v.w;
        } else {
            At[b] = (__bf16)0.f; At[b+1] = (__bf16)0.f;
            At[b+2] = (__bf16)0.f; At[b+3] = (__bf16)0.f;
        }
    }
    __syncthreads();

    const int w = tid >> 5, lane = tid & 31;
    const int mrow  = lane & 15;
    const int koff  = (lane >> 4) << 3;     // 0 or 8
    const int mbase = (lane >> 4) << 3;     // C rows: r + mbase

    for (int nt = 0; nt < 4; ++nt) {
        const int n = w * 64 + nt * 16 + (lane & 15);
        v8f acc0 = {0.f,0.f,0.f,0.f,0.f,0.f,0.f,0.f};
        v8f acc1 = {0.f,0.f,0.f,0.f,0.f,0.f,0.f,0.f};
#pragma unroll
        for (int ks = 0; ks < 4; ++ks) {
            const int kb = ks * 32 + koff;
            v16bf b  = frag16(Wb + (size_t)n * Dk + kb);
            v16bf a0 = frag16(At + mrow * Dk + kb);
            v16bf a1 = frag16(At + (16 + mrow) * Dk + kb);
            acc0 = __builtin_amdgcn_wmma_f32_16x16x32_bf16(
                       false, a0, false, b, (short)0, acc0, false, false);
            acc1 = __builtin_amdgcn_wmma_f32_16x16x32_bf16(
                       false, a1, false, b, (short)0, acc1, false, false);
        }
        // BN statistics: per-column sum & sumsq over this 32-row block.
        float s = 0.f, sq = 0.f;
#pragma unroll
        for (int r = 0; r < 8; ++r) {
            float v0 = acc0[r], v1 = acc1[r];
            s += v0 + v1; sq += v0 * v0 + v1 * v1;
        }
        s  += __shfl_xor(s, 16);
        sq += __shfl_xor(sq, 16);
        if (lane < 16) { atomicAdd(&colsum[n], s); atomicAdd(&colsq[n], sq); }
#pragma unroll
        for (int r = 0; r < 8; ++r) {
            int row = row0 + mbase + r;
            if (row < Nn)      z1[(size_t)row * FH + n] = (__bf16)acc0[r];
            if (row + 16 < Nn) z1[(size_t)(row + 16) * FH + n] = (__bf16)acc1[r];
        }
    }
}

// --------------------------- BN finalize (512 ch) ---------------------------
__global__ void k_bnfin(const float* __restrict__ colsum, const float* __restrict__ colsq,
                        const float* __restrict__ g, const float* __restrict__ b,
                        float* __restrict__ bnsc, float* __restrict__ bnsh, float Nf) {
    int i = blockIdx.x * blockDim.x + threadIdx.x;
    if (i >= FH) return;
    float mean = colsum[i] / Nf;
    float var  = colsq[i] / Nf - mean * mean;
    float sc   = g[i] * rsqrtf(var + BN_EPS);
    bnsc[i] = sc;
    bnsh[i] = b[i] - mean * sc;
}

// ------------------------- GEMM2: [N,512] x [512,128] -----------------------
// Block owns 32 rows; wave w owns N-tile w*16 (8 waves = 128 cols), 2 M-tiles.
// K loop 16 x 32; BN + SELU applied while staging A; bias + per-graph sums
// fused in the epilogue.
__global__ void __launch_bounds__(256)
k_gemm2(const __bf16* __restrict__ z1, const float* __restrict__ bnsc,
        const float* __restrict__ bnsh, const __bf16* __restrict__ Wb /*[128][512]*/,
        const float* __restrict__ bias, const int* __restrict__ batch,
        float* __restrict__ z2, float* __restrict__ gsum, int Nn) {
    __shared__ __bf16 At[32 * FH];   // 32 KB, row-major [32][512]
    const int tid  = threadIdx.x;
    const int row0 = blockIdx.x * 32;

    // Stage A tile: 16384 bf16, 64 per thread, apply BN + SELU.
    {
        const int base0 = tid * 64;
        const int r = base0 >> 9;               // 512 elems per row; 8 thr/row
        const bool ok = (row0 + r) < Nn;
        for (int j = 0; j < 64; j += 8) {
            const int base = base0 + j;
            const int c = base & (FH - 1);
            if (ok) {
                v8bf v = *(const v8bf*)(z1 + (size_t)(row0 + r) * FH + c);
#pragma unroll
                for (int u = 0; u < 8; ++u) {
                    float f = (float)v[u];
                    f = f * bnsc[c + u] + bnsh[c + u];
                    At[base + u] = (__bf16)selu_f(f);
                }
            } else {
#pragma unroll
                for (int u = 0; u < 8; ++u) At[base + u] = (__bf16)0.f;
            }
        }
    }
    __syncthreads();

    const int w = tid >> 5, lane = tid & 31;
    const int mrow  = lane & 15;
    const int koff  = (lane >> 4) << 3;
    const int mbase = (lane >> 4) << 3;
    const int n = w * 16 + (lane & 15);

    v8f acc0 = {0.f,0.f,0.f,0.f,0.f,0.f,0.f,0.f};
    v8f acc1 = {0.f,0.f,0.f,0.f,0.f,0.f,0.f,0.f};
#pragma unroll
    for (int ks = 0; ks < 16; ++ks) {
        const int kb = ks * 32 + koff;
        v16bf b  = frag16(Wb + (size_t)n * FH + kb);
        v16bf a0 = frag16(At + mrow * FH + kb);
        v16bf a1 = frag16(At + (16 + mrow) * FH + kb);
        acc0 = __builtin_amdgcn_wmma_f32_16x16x32_bf16(
                   false, a0, false, b, (short)0, acc0, false, false);
        acc1 = __builtin_amdgcn_wmma_f32_16x16x32_bf16(
                   false, a1, false, b, (short)0, acc1, false, false);
    }
    const float bi = bias[n];
#pragma unroll
    for (int r = 0; r < 8; ++r) {
        int row = row0 + mbase + r;
        if (row < Nn) {
            float v = acc0[r] + bi;
            z2[(size_t)row * Dk + n] = v;
            atomicAdd(&gsum[batch[row] * Dk + n], v);
        }
        if (row + 16 < Nn) {
            float v = acc1[r] + bi;
            z2[(size_t)(row + 16) * Dk + n] = v;
            atomicAdd(&gsum[batch[row + 16] * Dk + n], v);
        }
    }
}

// --------------------------- GraphNorm pass 1 -------------------------------
__global__ void k_gnvar(float* __restrict__ z2, const float* __restrict__ gsum,
                        const float* __restrict__ cnt, const float* __restrict__ gna,
                        const int* __restrict__ batch, float* __restrict__ gvar, int Nn) {
    int idx = blockIdx.x * blockDim.x + threadIdx.x;
    if (idx >= Nn * Dk) return;
    int i = idx >> 7, f = idx & (Dk - 1);
    int g = batch[i];
    float c = fmaxf(cnt[g], 1.0f);
    float sub = z2[idx] - gna[f] * (gsum[g * Dk + f] / c);
    z2[idx] = sub;
    atomicAdd(&gvar[g * Dk + f], sub * sub);
}

// --------------------------- GraphNorm pass 2 -------------------------------
__global__ void k_gnfin(const float* __restrict__ z2, const float* __restrict__ gvar,
                        const float* __restrict__ cnt, const float* __restrict__ gng,
                        const float* __restrict__ gnb, const int* __restrict__ batch,
                        float* __restrict__ h, float* __restrict__ pooled,
                        int layer, int Nn) {
    int idx = blockIdx.x * blockDim.x + threadIdx.x;
    if (idx >= Nn * Dk) return;
    int i = idx >> 7, f = idx & (Dk - 1);
    int g = batch[i];
    float c = fmaxf(cnt[g], 1.0f);
    float v = gng[f] * z2[idx] * rsqrtf(gvar[g * Dk + f] / c + GN_EPS) + gnb[f];
    v = selu_f(v);
    h[idx] = v;
    atomicAdd(&pooled[g * FH + layer * Dk + f], v);
}

__global__ void k_pooldiv(float* __restrict__ pooled, const float* __restrict__ cnt, int n) {
    int i = blockIdx.x * blockDim.x + threadIdx.x;
    if (i < n) pooled[i] /= fmaxf(cnt[i >> 9], 1.0f);
}

// --------------------------------- launcher ---------------------------------

extern "C" void kernel_launch(void* const* d_in, const int* in_sizes, int n_in,
                              void* d_out, int out_size, void* d_ws, size_t ws_size,
                              hipStream_t stream) {
    const float* x    = (const float*)d_in[0];
    const float* W1   = (const float*)d_in[1];   // [4,512,128]
    const float* bn_g = (const float*)d_in[2];
    const float* bn_b = (const float*)d_in[3];
    const float* W2   = (const float*)d_in[4];   // [4,128,512]
    const float* b2   = (const float*)d_in[5];
    const float* gn_g = (const float*)d_in[6];
    const float* gn_b = (const float*)d_in[7];
    const float* gn_a = (const float*)d_in[8];
    const int*   ei   = (const int*)d_in[9];     // [2,E]
    const int*   batch= (const int*)d_in[10];    // [N]

    const int N = in_sizes[0] / Dk;
    const int E = in_sizes[9] / 2;
    float* out = (float*)d_out;                  // [256,512]

    // Workspace layout (all offsets 16B-aligned).
    char* ws = (char*)d_ws;
    size_t off = 0;
    float*  h    = (float*)(ws + off);  off += (size_t)N * Dk * 4;
    float*  zin  = (float*)(ws + off);  off += (size_t)N * Dk * 4;
    __bf16* z1   = (__bf16*)(ws + off); off += (size_t)N * FH * 2;
    float*  z2   = (float*)(ws + off);  off += (size_t)N * Dk * 4;
    __bf16* W1b  = (__bf16*)(ws + off); off += (size_t)4 * FH * Dk * 2;
    __bf16* W2b  = (__bf16*)(ws + off); off += (size_t)4 * Dk * FH * 2;
    // per-layer zeroed region (contiguous): colsum, colsq, gsum, gvar
    float*  colsum = (float*)(ws + off); off += FH * 4;
    float*  colsq  = (float*)(ws + off); off += FH * 4;
    float*  gsum   = (float*)(ws + off); off += (size_t)NG * Dk * 4;
    float*  gvar   = (float*)(ws + off); off += (size_t)NG * Dk * 4;
    float*  bnsc   = (float*)(ws + off); off += FH * 4;
    float*  bnsh   = (float*)(ws + off); off += FH * 4;
    float*  cnt    = (float*)(ws + off); off += NG * 4;

    const int T = 256;
    const int ND = N * Dk;
    const int rowBlocks = (N + 31) / 32;
    const int zlayerN = FH + FH + NG * Dk + NG * Dk;   // colsum..gvar

    // One-time prep.
    k_zero<<<(out_size + T - 1) / T, T, 0, stream>>>(out, out_size);
    k_zero<<<(NG + T - 1) / T, T, 0, stream>>>(cnt, NG);
    k_count<<<(N + T - 1) / T, T, 0, stream>>>(batch, cnt, N);
    k_cvt_bf16<<<(4 * FH * Dk + T - 1) / T, T, 0, stream>>>(W1, W1b, 4 * FH * Dk);
    k_cvt_bf16<<<(4 * Dk * FH + T - 1) / T, T, 0, stream>>>(W2, W2b, 4 * Dk * FH);

    const float* hin = x;
    for (int l = 0; l < 4; ++l) {
        k_zero<<<(zlayerN + T - 1) / T, T, 0, stream>>>(colsum, zlayerN);
        // zin = h, then += scatter of neighbors.
        k_copy<<<(ND + T - 1) / T, T, 0, stream>>>(hin, zin, ND);
        k_scatter<<<(E * 32 + T - 1) / T, T, 0, stream>>>(hin, ei, zin, E);
        // z1 = zin @ W1^T (+ BN stats)
        k_gemm1<<<rowBlocks, T, 0, stream>>>(zin, W1b + (size_t)l * FH * Dk,
                                             z1, colsum, colsq, N);
        k_bnfin<<<(FH + T - 1) / T, T, 0, stream>>>(colsum, colsq,
                                                    bn_g + l * FH, bn_b + l * FH,
                                                    bnsc, bnsh, (float)N);
        // z2 = SELU(BN(z1)) @ W2^T + b2 (+ per-graph sums)
        k_gemm2<<<rowBlocks, T, 0, stream>>>(z1, bnsc, bnsh,
                                             W2b + (size_t)l * Dk * FH,
                                             b2 + l * Dk, batch, z2, gsum, N);
        // GraphNorm + SELU -> h, pooled accumulation.
        k_gnvar<<<(ND + T - 1) / T, T, 0, stream>>>(z2, gsum, cnt, gn_a + l * Dk,
                                                    batch, gvar, N);
        k_gnfin<<<(ND + T - 1) / T, T, 0, stream>>>(z2, gvar, cnt, gn_g + l * Dk,
                                                    gn_b + l * Dk, batch, h, out, l, N);
        hin = h;
    }
    k_pooldiv<<<(out_size + T - 1) / T, T, 0, stream>>>(out, cnt, out_size);
}
